// PureGAT_30966714204820
// MI455X (gfx1250) — matrix operand order repaired and checked by reading, not compiled, not run on hardware
//
#include <hip/hip_runtime.h>

typedef __attribute__((ext_vector_type(16))) __bf16 v16bf;
typedef __attribute__((ext_vector_type(8)))  float  v8f;

#define NEG_SLOPE 0.2f
#define NUM_GRAPHS 64

static __device__ __forceinline__ void atomicMaxFloat(float* addr, float val) {
  // Valid for any sign mix when the slot is initialized to -FLT_MAX:
  // nonnegative floats order like signed ints, negative floats order inversely as uints.
  if (val >= 0.0f) atomicMax((int*)addr, __float_as_int(val));
  else             atomicMin((unsigned int*)addr, __float_as_uint(val));
}

// ---------------------------------------------------------------------------
// H[N, OUTW(sub)] = X[N, K] @ Wm[OUTW, K]^T   via v_wmma_f32_16x16x32_bf16
// Block = 128 threads = 4 waves; each wave: one 16-row M tile x NTILES N-tiles.
// gridDim.y covers OUTW in chunks of NTILES*16 columns.
// ---------------------------------------------------------------------------
template<int K, int OUTW, int NTILES>
__global__ __launch_bounds__(128) void gemm_bf16_wmma(const float* __restrict__ X,
                                                      const float* __restrict__ Wm,
                                                      float* __restrict__ H, int N) {
  const int lane  = threadIdx.x & 31;
  const int wave  = threadIdx.x >> 5;
  const int l16   = lane & 15;
  const int hi    = lane >> 4;              // 0: K 0..7,16..23; 1: K 8..15,24..31
  const int mbase = (blockIdx.x * 4 + wave) * 16;
  const int nb0   = blockIdx.y * (NTILES * 16);

  v8f acc[NTILES] = {};

  const int  row   = mbase + l16;
  const bool rowok = (row < N);
  const int  kbA   = hi * 8;                // A: lanes 0-15 K 0..7 & 16..23; lanes 16-31 K 8..15 & 24..31
  const int  kbB   = hi * 8;                // B: lanes 0-15 K 0..7 & 16..23; lanes 16-31 K 8..15 & 24..31

  for (int k0 = 0; k0 < K; k0 += 32) {
    v16bf a;
    if (rowok) {
#pragma unroll
      for (int j = 0; j < 8; ++j) {
        a[j]     = (__bf16)X[(size_t)row * K + k0 + kbA + j];
        a[j + 8] = (__bf16)X[(size_t)row * K + k0 + 16 + kbA + j];
      }
    } else {
#pragma unroll
      for (int j = 0; j < 16; ++j) a[j] = (__bf16)0.0f;
    }
#pragma unroll
    for (int t = 0; t < NTILES; ++t) {
      const int n = nb0 + t * 16 + l16;     // B column = W row
      v16bf b;
#pragma unroll
      for (int j = 0; j < 16; ++j) b[j] = (__bf16)Wm[(size_t)n * K + k0 + kbB + j + (j & 8)];
      acc[t] = __builtin_amdgcn_wmma_f32_16x16x32_bf16(false, a, false, b,
                                                       (short)0, acc[t], false, false);
    }
  }
  // C/D layout: VGPR r, lanes 0-15 -> (M=r, N=l16); lanes 16-31 -> (M=r+8, N=l16)
#pragma unroll
  for (int t = 0; t < NTILES; ++t) {
    const int n = nb0 + t * 16 + l16;
#pragma unroll
    for (int r = 0; r < 8; ++r) {
      const int m = mbase + r + hi * 8;
      if (m < N) H[(size_t)m * OUTW + n] = acc[t][r];
    }
  }
}

// ---------------------------------------------------------------------------
// Per-node attention scores: a_src[n,h] = sum_d h[n,h*D+d]*att_src[h,d]
// ---------------------------------------------------------------------------
template<int HH, int D>
__global__ void attn_scores_k(const float* __restrict__ Hm, const float* __restrict__ att_s,
                              const float* __restrict__ att_d, float* __restrict__ as_out,
                              float* __restrict__ ad_out, int N) {
  int i = blockIdx.x * blockDim.x + threadIdx.x;   // i = n*HH + h
  if (i >= N * HH) return;
  int n = i / HH, h = i % HH;
  const float* hp = Hm + (size_t)n * (HH * D) + h * D;
  float ss = 0.f, sd = 0.f;
#pragma unroll
  for (int d = 0; d < D; ++d) {
    float v = hp[d];
    ss += v * att_s[h * D + d];
    sd += v * att_d[h * D + d];
  }
  as_out[i] = ss;
  ad_out[i] = sd;
}

__global__ void fill_f32(float* __restrict__ p, float v, long long n) {
  long long i = (long long)blockIdx.x * blockDim.x + threadIdx.x;
  if (i < n) p[i] = v;
}

// ---------------------------------------------------------------------------
// Edge passes. Edge list = [E real edges] ++ [N self loops].
// Pass 1: e = leakyrelu(a_src[s]+a_dst[d]); alpha_buf <- e; m[d] <- max
// Pass 2: t = exp(alpha_buf - m[d]); alpha_buf <- t; ssum[d] += t
// Pass 3 (agg): alpha = alpha_buf / (ssum[d]+eps); out[d, f0..f0+3] += alpha*feat[s, f0..f0+3]
// ---------------------------------------------------------------------------
template<int HH>
__global__ void edge_max_k(const long long* __restrict__ ei, int E, int N,
                           const float* __restrict__ asc, const float* __restrict__ adc,
                           float* __restrict__ m, float* __restrict__ alpha) {
  long long idx = (long long)blockIdx.x * blockDim.x + threadIdx.x;
  long long tot = (long long)(E + N) * HH;
  if (idx >= tot) return;
  int e = (int)(idx / HH), h = (int)(idx % HH);
  int s, d;
  if (e < E) { s = (int)ei[e]; d = (int)ei[E + e]; } else { s = d = e - E; }
  float v = asc[(size_t)s * HH + h] + adc[(size_t)d * HH + h];
  v = (v >= 0.f) ? v : NEG_SLOPE * v;
  alpha[idx] = v;
  atomicMaxFloat(&m[(size_t)d * HH + h], v);
}

template<int HH>
__global__ void edge_exp_k(const long long* __restrict__ ei, int E, int N,
                           const float* __restrict__ m, float* __restrict__ ssum,
                           float* __restrict__ alpha) {
  long long idx = (long long)blockIdx.x * blockDim.x + threadIdx.x;
  long long tot = (long long)(E + N) * HH;
  if (idx >= tot) return;
  int e = (int)(idx / HH), h = (int)(idx % HH);
  int d = (e < E) ? (int)ei[E + e] : (e - E);
  float t = expf(alpha[idx] - m[(size_t)d * HH + h]);
  alpha[idx] = t;
  atomicAdd(&ssum[(size_t)d * HH + h], t);
}

template<int HH, int D>
__global__ void edge_agg_k(const long long* __restrict__ ei, int E, int N,
                           const float* __restrict__ ssum, const float* __restrict__ alpha,
                           const float* __restrict__ feat, float* __restrict__ out) {
  const int F = HH * D;
  const int C = F / 4;                       // 4 features per thread
  long long idx = (long long)blockIdx.x * blockDim.x + threadIdx.x;
  long long tot = (long long)(E + N) * C;
  if (idx >= tot) return;
  int e  = (int)(idx / C);
  int c  = (int)(idx % C);
  int f0 = c * 4;
  int h  = f0 / D;
  int s, d;
  if (e < E) { s = (int)ei[e]; d = (int)ei[E + e]; } else { s = d = e - E; }
  float a = alpha[(size_t)e * HH + h] / (ssum[(size_t)d * HH + h] + 1e-16f);
  const float4 v = *(const float4*)(feat + (size_t)s * F + f0);
  float* op = out + (size_t)d * F + f0;
  atomicAdd(op + 0, a * v.x);
  atomicAdd(op + 1, a * v.y);
  atomicAdd(op + 2, a * v.z);
  atomicAdd(op + 3, a * v.w);
}

__global__ void bias_relu_k(float* __restrict__ x, const float* __restrict__ b,
                            long long n, int F) {
  long long i = (long long)blockIdx.x * blockDim.x + threadIdx.x;
  if (i >= n) return;
  int f = (int)(i % F);
  float v = x[i] + b[f];
  x[i] = (v > 0.f) ? v : 0.f;
}

// ---------------------------------------------------------------------------
// Mean pool per graph, then FC head
// ---------------------------------------------------------------------------
__global__ void pool_sum_k(const float* __restrict__ x1, const float* __restrict__ x2,
                           const long long* __restrict__ batch, float* __restrict__ pool,
                           int N) {
  long long i = (long long)blockIdx.x * blockDim.x + threadIdx.x;
  if (i >= (long long)N * 320) return;
  int n = (int)(i / 320), f = (int)(i % 320);
  int g = (int)batch[n];
  float v = (f < 64) ? x1[(size_t)n * 64 + f] : x2[(size_t)n * 256 + (f - 64)];
  atomicAdd(&pool[(size_t)g * 320 + f], v);
}

__global__ void pool_cnt_k(const long long* __restrict__ batch, float* __restrict__ cnt, int N) {
  int i = blockIdx.x * blockDim.x + threadIdx.x;
  if (i >= N) return;
  atomicAdd(&cnt[(int)batch[i]], 1.0f);
}

__global__ void fc1_k(const float* __restrict__ pool, const float* __restrict__ cnt,
                      const float* __restrict__ w, const float* __restrict__ b,
                      float* __restrict__ hout) {
  int i = blockIdx.x * blockDim.x + threadIdx.x;   // 64*128
  if (i >= NUM_GRAPHS * 128) return;
  int g = i / 128, j = i % 128;
  float inv = 1.0f / fmaxf(cnt[g], 1.0f);
  float acc = b[j];
  for (int k = 0; k < 320; ++k) acc += (pool[g * 320 + k] * inv) * w[j * 320 + k];
  hout[i] = fmaxf(acc, 0.0f);
}

__global__ void fc2_k(const float* __restrict__ h, const float* __restrict__ w,
                      const float* __restrict__ b, float* __restrict__ out) {
  int g = blockIdx.x * blockDim.x + threadIdx.x;
  if (g >= NUM_GRAPHS) return;
  float acc = b[0];
  for (int j = 0; j < 128; ++j) acc += h[g * 128 + j] * w[j];
  out[g] = acc;
}

// ---------------------------------------------------------------------------
extern "C" void kernel_launch(void* const* d_in, const int* in_sizes, int n_in,
                              void* d_out, int out_size, void* d_ws, size_t ws_size,
                              hipStream_t stream) {
  const float*     x     = (const float*)d_in[0];
  const long long* ei    = (const long long*)d_in[1];
  const long long* batch = (const long long*)d_in[2];
  const float* W1   = (const float*)d_in[3];
  const float* as1w = (const float*)d_in[4];
  const float* ad1w = (const float*)d_in[5];
  const float* b1   = (const float*)d_in[6];
  const float* W2   = (const float*)d_in[7];
  const float* as2w = (const float*)d_in[8];
  const float* ad2w = (const float*)d_in[9];
  const float* b2   = (const float*)d_in[10];
  const float* fc1w = (const float*)d_in[11];
  const float* fc1b = (const float*)d_in[12];
  const float* fc2w = (const float*)d_in[13];
  const float* fc2b = (const float*)d_in[14];
  float* out = (float*)d_out;

  const int N = in_sizes[0] / 128;
  const int E = in_sizes[1] / 2;
  const long long Etot = (long long)E + N;

  // Workspace carve-up (all f32, every offset a multiple of 4 elements)
  float* ws = (float*)d_ws;
  size_t off = 0;
  float* h1   = ws + off; off += (size_t)N * 64;
  float* a_s1 = ws + off; off += (size_t)N * 4;
  float* a_d1 = ws + off; off += (size_t)N * 4;
  float* m1   = ws + off; off += (size_t)N * 4;
  float* s1   = ws + off; off += (size_t)N * 4;
  float* x1   = ws + off; off += (size_t)N * 64;
  float* h2   = ws + off; off += (size_t)N * 256;
  float* a_s2 = ws + off; off += (size_t)N * 8;
  float* a_d2 = ws + off; off += (size_t)N * 8;
  float* m2   = ws + off; off += (size_t)N * 8;
  float* s2   = ws + off; off += (size_t)N * 8;
  float* x2   = ws + off; off += (size_t)N * 256;
  float* alpha = ws + off; off += (size_t)Etot * 8;      // reused by both layers
  float* pool = ws + off; off += (size_t)NUM_GRAPHS * 320;
  float* cnt  = ws + off; off += 64;
  float* fch  = ws + off; off += (size_t)NUM_GRAPHS * 128;
  (void)ws_size; (void)n_in; (void)out_size;

  const int T = 256;
  auto blocks = [](long long n, int t) { return (unsigned)((n + t - 1) / t); };
  const unsigned mblk = (unsigned)((N + 63) / 64);

  // ---- Layer 1: h1 = x @ W1^T  (K=128, OUTW=64), H=4, D=16 ----
  gemm_bf16_wmma<128, 64, 4><<<dim3(mblk, 1), 128, 0, stream>>>(x, W1, h1, N);
  attn_scores_k<4, 16><<<blocks((long long)N * 4, T), T, 0, stream>>>(h1, as1w, ad1w, a_s1, a_d1, N);

  fill_f32<<<blocks((long long)N * 4, T), T, 0, stream>>>(m1, -3.402823466e38f, (long long)N * 4);
  fill_f32<<<blocks((long long)N * 4, T), T, 0, stream>>>(s1, 0.0f, (long long)N * 4);
  fill_f32<<<blocks((long long)N * 64, T), T, 0, stream>>>(x1, 0.0f, (long long)N * 64);

  edge_max_k<4><<<blocks(Etot * 4, T), T, 0, stream>>>(ei, E, N, a_s1, a_d1, m1, alpha);
  edge_exp_k<4><<<blocks(Etot * 4, T), T, 0, stream>>>(ei, E, N, m1, s1, alpha);
  edge_agg_k<4, 16><<<blocks(Etot * 16, T), T, 0, stream>>>(ei, E, N, s1, alpha, h1, x1);
  bias_relu_k<<<blocks((long long)N * 64, T), T, 0, stream>>>(x1, b1, (long long)N * 64, 64);

  // ---- Layer 2: h2 = x1 @ W2^T  (K=64, OUTW=256), H=8, D=32 ----
  gemm_bf16_wmma<64, 256, 4><<<dim3(mblk, 4), 128, 0, stream>>>(x1, W2, h2, N);
  attn_scores_k<8, 32><<<blocks((long long)N * 8, T), T, 0, stream>>>(h2, as2w, ad2w, a_s2, a_d2, N);

  fill_f32<<<blocks((long long)N * 8, T), T, 0, stream>>>(m2, -3.402823466e38f, (long long)N * 8);
  fill_f32<<<blocks((long long)N * 8, T), T, 0, stream>>>(s2, 0.0f, (long long)N * 8);
  fill_f32<<<blocks((long long)N * 256, T), T, 0, stream>>>(x2, 0.0f, (long long)N * 256);

  edge_max_k<8><<<blocks(Etot * 8, T), T, 0, stream>>>(ei, E, N, a_s2, a_d2, m2, alpha);
  edge_exp_k<8><<<blocks(Etot * 8, T), T, 0, stream>>>(ei, E, N, m2, s2, alpha);
  edge_agg_k<8, 32><<<blocks(Etot * 64, T), T, 0, stream>>>(ei, E, N, s2, alpha, h2, x2);
  bias_relu_k<<<blocks((long long)N * 256, T), T, 0, stream>>>(x2, b2, (long long)N * 256, 256);

  // ---- Pool + FC head ----
  fill_f32<<<blocks((long long)NUM_GRAPHS * 320 + 64, T), T, 0, stream>>>(pool, 0.0f,
                                                     (long long)NUM_GRAPHS * 320 + 64);
  pool_sum_k<<<blocks((long long)N * 320, T), T, 0, stream>>>(x1, x2, batch, pool, N);
  pool_cnt_k<<<blocks(N, T), T, 0, stream>>>(batch, cnt, N);
  fc1_k<<<blocks(NUM_GRAPHS * 128, T), T, 0, stream>>>(pool, cnt, fc1w, fc1b, fch);
  fc2_k<<<1, 64, 0, stream>>>(fch, fc2w, fc2b, out);
}
